// NodeGIN_70944269795973
// MI455X (gfx1250) — matrix operand order repaired and checked by reading, not compiled, not run on hardware
//
#include <hip/hip_runtime.h>

typedef __attribute__((ext_vector_type(16))) _Float16 v16h;
typedef __attribute__((ext_vector_type(8)))  _Float16 v8h;
typedef __attribute__((ext_vector_type(8)))  float    v8f;

#define HIDDEN 20

__device__ __forceinline__ v8f wmma16x16x32(v16h a, v16h b, v8f c) {
  // D = A(16x32 f16) * B(32x16 f16) + C(16x16 f32)
  return __builtin_amdgcn_wmma_f32_16x16x32_f16(
      /*neg_a=*/false, a, /*neg_b=*/false, b,
      /*c_mod=*/(short)0, c, /*reuse_a=*/false, /*reuse_b=*/false);
}

// ---------------------------------------------------------------------------
// Edge scatter, layer 1 (64 features). 16 lanes cooperate on one edge so the
// row gather of x[src] is one coalesced 256B transaction; atomics go to
// consecutive addresses of agg[dst] (L2-resident working set).
// ---------------------------------------------------------------------------
__global__ void scatter64_kernel(const float* __restrict__ x,
                                 const int* __restrict__ ei,
                                 const float* __restrict__ ew,
                                 float* __restrict__ agg, int E) {
  long long t = (long long)blockIdx.x * blockDim.x + threadIdx.x;
  int e = (int)(t >> 4);
  if (e >= E) return;
  int q = (int)(t & 15);
  int s = ei[e];
  int d = ei[E + e];
  float w = ew[e];
  const float4 v = *reinterpret_cast<const float4*>(x + (size_t)s * 64 + q * 4);
  float* p = agg + (size_t)d * 64 + q * 4;
  atomicAdd(p + 0, v.x * w);
  atomicAdd(p + 1, v.y * w);
  atomicAdd(p + 2, v.z * w);
  atomicAdd(p + 3, v.w * w);
}

// Edge scatter, 20-feature layers: thread per (edge, feature).
// Destination buffer is stride-32 (zero-padded) so the WMMA A-loads need no
// predication.
__global__ void scatter20_kernel(const float* __restrict__ h,
                                 const int* __restrict__ ei,
                                 const float* __restrict__ ew,
                                 float* __restrict__ agg, int E) {
  long long t = (long long)blockIdx.x * blockDim.x + threadIdx.x;
  int e = (int)(t / HIDDEN);
  if (e >= E) return;
  int f = (int)(t - (long long)e * HIDDEN);
  int s = ei[e];
  int d = ei[E + e];
  float w = ew[e];
  atomicAdd(agg + (size_t)d * 32 + f, h[(size_t)s * HIDDEN + f] * w);
}

// ---------------------------------------------------------------------------
// Pre-pack weights into the exact per-lane WMMA B-fragment layout (f16,
// zero-padded for k >= DIN and n >= 20) and zero-padded 32-entry biases.
// Fragment flat index: ((kc*2 + nt)*32 + lane)*16 + j, element = W[k][n] with
// k = kc*32 + (j>>3)*16 + (lane>>4)*8 + (j&7), n = nt*16 + (lane&15).
// ---------------------------------------------------------------------------
template <int DIN, int KCH>
__global__ void prep_weights_kernel(const float* __restrict__ W1,
                                    const float* __restrict__ B1,
                                    const float* __restrict__ W2,
                                    const float* __restrict__ B2,
                                    _Float16* __restrict__ w1f,
                                    _Float16* __restrict__ w2f,
                                    float* __restrict__ b1p,
                                    float* __restrict__ b2p) {
  const int NW1 = KCH * 2 * 32 * 16;
  const int NW2 = 2 * 32 * 16;
  for (int i = threadIdx.x; i < NW1; i += blockDim.x) {
    int j    = i & 15;
    int lane = (i >> 4) & 31;
    int nt   = (i >> 9) & 1;
    int kc   = i >> 10;
    int k = kc * 32 + (j >> 3) * 16 + (lane >> 4) * 8 + (j & 7);
    int n = nt * 16 + (lane & 15);
    w1f[i] = (k < DIN && n < HIDDEN) ? (_Float16)W1[k * HIDDEN + n] : (_Float16)0.f;
  }
  for (int i = threadIdx.x; i < NW2; i += blockDim.x) {
    int j    = i & 15;
    int lane = (i >> 4) & 31;
    int nt   = (i >> 9) & 1;
    int k = (j >> 3) * 16 + (lane >> 4) * 8 + (j & 7);
    int n = nt * 16 + (lane & 15);
    w2f[i] = (k < HIDDEN && n < HIDDEN) ? (_Float16)W2[k * HIDDEN + n] : (_Float16)0.f;
  }
  if (threadIdx.x < 32) {
    b1p[threadIdx.x] = (threadIdx.x < HIDDEN) ? B1[threadIdx.x] : 0.f;
    b2p[threadIdx.x] = (threadIdx.x < HIDDEN) ? B2[threadIdx.x] : 0.f;
  }
}

// ---------------------------------------------------------------------------
// Fused per-node 2-layer MLP: h = relu(relu(agg@W1+b1)@W2+b2), via WMMA.
// One wave computes a 16-node tile. DSTRIDE = padded input row stride (64 or
// 32); input padding columns are zero, so all loads are unconditional.
// z is staged through per-wave LDS to convert from C/D layout to A layout
// (same-wave LDS ops are in-order).
// ---------------------------------------------------------------------------
template <int DSTRIDE, int KCH>
__global__ void __launch_bounds__(256) mlp_wmma_kernel(
    const float* __restrict__ agg,
    const _Float16* __restrict__ w1f, const float* __restrict__ b1p,
    const _Float16* __restrict__ w2f, const float* __restrict__ b2p,
    float* __restrict__ out, int n_tiles) {
  __shared__ _Float16 zbuf[8][16 * 32];  // 8 waves x (16x32) f16 = 8KB
  const int wave = threadIdx.x >> 5;
  const int lane = threadIdx.x & 31;
  const int tile = blockIdx.x * 8 + wave;
  if (tile >= n_tiles) return;          // whole-wave exit: EXEC stays all-1s
  const int row    = lane & 15;         // M row of A / N col of B,C,D
  const int half   = lane >> 4;
  const int khalfA = half * 8;          // A layout: K offset for this half
  const int n1     = 16 + row;
  const float* arow = agg + (size_t)(tile * 16 + row) * DSTRIDE;

  v8f c0 = {0.f, 0.f, 0.f, 0.f, 0.f, 0.f, 0.f, 0.f};
  v8f c1 = c0;

  // -------- GEMM 1: z = agg(16xK) @ W1, K in chunks of 32 --------
  #pragma unroll
  for (int kc = 0; kc < KCH; ++kc) {
    float av[16];
    ((float4*)av)[0] = *(const float4*)(arow + kc * 32 + khalfA);
    ((float4*)av)[1] = *(const float4*)(arow + kc * 32 + khalfA + 4);
    ((float4*)av)[2] = *(const float4*)(arow + kc * 32 + 16 + khalfA);
    ((float4*)av)[3] = *(const float4*)(arow + kc * 32 + 16 + khalfA + 4);
    v16h a;
    #pragma unroll
    for (int j = 0; j < 16; ++j) a[j] = (_Float16)av[j];

    const v16h b0  = *(const v16h*)(w1f + ((size_t)(kc * 2 + 0) * 32 + lane) * 16);
    const v16h b1v = *(const v16h*)(w1f + ((size_t)(kc * 2 + 1) * 32 + lane) * 16);
    c0 = wmma16x16x32(a, b0, c0);
    c1 = wmma16x16x32(a, b1v, c1);
  }

  // -------- bias + relu, stage z (padding cols are relu(0+0)=0) --------
  _Float16* z = zbuf[wave];
  {
    float bias0 = b1p[row];
    float bias1 = b1p[n1];           // zero for n1 >= 20
    #pragma unroll
    for (int r = 0; r < 8; ++r) {
      int m = r + half * 8;
      z[m * 32 + row] = (_Float16)fmaxf(c0[r] + bias0, 0.f);
      z[m * 32 + n1]  = (_Float16)fmaxf(c1[r] + bias1, 0.f);
    }
  }

  // -------- GEMM 2: h = z(16x32, K=20 padded) @ W2 --------
  v16h a2;
  {
    v8h lo = *(const v8h*)(z + row * 32 + khalfA);
    v8h hi = *(const v8h*)(z + row * 32 + 16 + khalfA);
    #pragma unroll
    for (int j = 0; j < 8; ++j) { a2[j] = lo[j]; a2[8 + j] = hi[j]; }
  }
  const v16h wb0 = *(const v16h*)(w2f + ((size_t)0 * 32 + lane) * 16);
  const v16h wb1 = *(const v16h*)(w2f + ((size_t)1 * 32 + lane) * 16);
  v8f d0 = {0.f, 0.f, 0.f, 0.f, 0.f, 0.f, 0.f, 0.f};
  v8f d1 = d0;
  d0 = wmma16x16x32(a2, wb0, d0);
  d1 = wmma16x16x32(a2, wb1, d1);

  // -------- bias + relu + store h (N x 20, f32) --------
  {
    float bias0 = b2p[row];
    float bias1 = b2p[n1];
    #pragma unroll
    for (int r = 0; r < 8; ++r) {
      int m = r + half * 8;
      out[(size_t)(tile * 16 + m) * HIDDEN + row] = fmaxf(d0[r] + bias0, 0.f);
      if (n1 < HIDDEN) {
        out[(size_t)(tile * 16 + m) * HIDDEN + n1] = fmaxf(d1[r] + bias1, 0.f);
      }
    }
  }
}

// ---------------------------------------------------------------------------
// BatchNorm statistics: per-column sum / sumsq (grid-strided streams, one
// column per lane slot, two global atomics per thread at the end).
// ---------------------------------------------------------------------------
__global__ void bn_stats_kernel(const float* __restrict__ h,
                                float* __restrict__ sum, float* __restrict__ sq,
                                int N) {
  int t = blockIdx.x * blockDim.x + threadIdx.x;
  int c = t & 31;
  if (c >= HIDDEN) return;
  int s0 = t >> 5;
  int nstreams = (gridDim.x * blockDim.x) >> 5;
  float a = 0.f, b = 0.f;
  for (int n = s0; n < N; n += nstreams) {
    float v = h[(size_t)n * HIDDEN + c];
    a += v;
    b += v * v;
  }
  atomicAdd(&sum[c], a);
  atomicAdd(&sq[c], b);
}

__global__ void bn_apply_kernel(float* __restrict__ h,
                                const float* __restrict__ sum,
                                const float* __restrict__ sq,
                                const float* __restrict__ gamma,
                                const float* __restrict__ beta, int N) {
  long long t = (long long)blockIdx.x * blockDim.x + threadIdx.x;
  if (t >= (long long)N * HIDDEN) return;
  int c = (int)(t % HIDDEN);
  float invN = 1.f / (float)N;
  float mu  = sum[c] * invN;
  float var = sq[c] * invN - mu * mu;
  float inv = rsqrtf(var + 1e-5f);
  h[t] = gamma[c] * (h[t] - mu) * inv + beta[c];
}

// ---------------------------------------------------------------------------
// Final linear: out = concat(o1,o2,o3) @ wlin(60x10) + blin. Memory-bound;
// wlin/blin cached in LDS.
// ---------------------------------------------------------------------------
__global__ void final_linear_kernel(const float* __restrict__ o1,
                                    const float* __restrict__ o2,
                                    const float* __restrict__ o3,
                                    const float* __restrict__ wlin,
                                    const float* __restrict__ blin,
                                    float* __restrict__ out, int N) {
  __shared__ float w[60 * 10];
  __shared__ float b[10];
  for (int i = threadIdx.x; i < 600; i += blockDim.x) w[i] = wlin[i];
  if (threadIdx.x < 10) b[threadIdx.x] = blin[threadIdx.x];
  __syncthreads();
  int n = blockIdx.x * blockDim.x + threadIdx.x;
  if (n >= N) return;
  float e[60];
  #pragma unroll
  for (int j = 0; j < HIDDEN; ++j) {
    e[j]      = o1[(size_t)n * HIDDEN + j];
    e[20 + j] = o2[(size_t)n * HIDDEN + j];
    e[40 + j] = o3[(size_t)n * HIDDEN + j];
  }
  #pragma unroll
  for (int c = 0; c < 10; ++c) {
    float acc = b[c];
    #pragma unroll
    for (int j = 0; j < 60; ++j) acc += e[j] * w[j * 10 + c];
    out[(size_t)n * 10 + c] = acc;
  }
}

// ---------------------------------------------------------------------------
extern "C" void kernel_launch(void* const* d_in, const int* in_sizes, int n_in,
                              void* d_out, int out_size, void* d_ws, size_t ws_size,
                              hipStream_t stream) {
  const float* x    = (const float*)d_in[0];
  const int*   ei   = (const int*)d_in[1];
  const float* ew   = (const float*)d_in[2];
  const float* w1a  = (const float*)d_in[3];
  const float* b1a  = (const float*)d_in[4];
  const float* w2a  = (const float*)d_in[5];
  const float* b2a  = (const float*)d_in[6];
  const float* g1   = (const float*)d_in[7];
  const float* be1  = (const float*)d_in[8];
  const float* w1b  = (const float*)d_in[9];
  const float* b1b  = (const float*)d_in[10];
  const float* w2b  = (const float*)d_in[11];
  const float* b2b  = (const float*)d_in[12];
  const float* g2   = (const float*)d_in[13];
  const float* be2  = (const float*)d_in[14];
  const float* w1c  = (const float*)d_in[15];
  const float* b1c  = (const float*)d_in[16];
  const float* w2c  = (const float*)d_in[17];
  const float* b2c  = (const float*)d_in[18];
  const float* wlin = (const float*)d_in[19];
  const float* blin = (const float*)d_in[20];

  const int N = in_sizes[0] / 64;     // 100000
  const int E = in_sizes[2];          // 3200000
  const int tiles = N / 16;           // 6250 (N is a multiple of 16)

  float* ws    = (float*)d_ws;
  float* agg   = ws;                              // N*64 (reused as N*32)
  float* o1    = agg + (size_t)N * 64;            // N*20
  float* o2    = o1 + (size_t)N * HIDDEN;         // N*20
  float* o3    = o2 + (size_t)N * HIDDEN;         // N*20
  float* stats = o3 + (size_t)N * HIDDEN;         // 128 floats

  // f16 weight-fragment area (32B-aligned offsets), then padded biases.
  _Float16* fh = (_Float16*)(stats + 128);
  _Float16* w1fa = fh + 0;      // 2048 halfs (KCH=2)
  _Float16* w2fa = fh + 2048;   // 1024 halfs
  _Float16* w1fb = fh + 3072;   // 1024 halfs (KCH=1)
  _Float16* w2fb = fh + 4096;   // 1024 halfs
  _Float16* w1fc = fh + 5120;   // 1024 halfs
  _Float16* w2fc = fh + 6144;   // 1024 halfs
  float* bp = stats + 128 + (7168 / 2);           // 6 x 32 padded biases

  const dim3 blk(256);
  const int mlp_blocks   = (tiles + 7) / 8;
  const int sc64_blocks  = (int)(((long long)E * 16 + 255) / 256);
  const int sc20_blocks  = (int)(((long long)E * HIDDEN + 255) / 256);
  const int elem_blocks  = (int)(((long long)N * HIDDEN + 255) / 256);
  const int node_blocks  = (N + 255) / 256;

  hipMemsetAsync(stats, 0, 128 * sizeof(float), stream);
  prep_weights_kernel<64, 2><<<1, blk, 0, stream>>>(w1a, b1a, w2a, b2a,
                                                    w1fa, w2fa, bp + 0, bp + 32);
  prep_weights_kernel<20, 1><<<1, blk, 0, stream>>>(w1b, b1b, w2b, b2b,
                                                    w1fb, w2fb, bp + 64, bp + 96);
  prep_weights_kernel<20, 1><<<1, blk, 0, stream>>>(w1c, b1c, w2c, b2c,
                                                    w1fc, w2fc, bp + 128, bp + 160);

  // ---- layer 1 (64 -> 20 -> 20) + BN ----
  hipMemsetAsync(agg, 0, (size_t)N * 64 * sizeof(float), stream);
  scatter64_kernel<<<sc64_blocks, blk, 0, stream>>>(x, ei, ew, agg, E);
  mlp_wmma_kernel<64, 2><<<mlp_blocks, blk, 0, stream>>>(agg, w1fa, bp + 0,
                                                         w2fa, bp + 32, o1, tiles);
  bn_stats_kernel<<<128, blk, 0, stream>>>(o1, stats + 0, stats + 32, N);
  bn_apply_kernel<<<elem_blocks, blk, 0, stream>>>(o1, stats + 0, stats + 32, g1, be1, N);

  // ---- layer 2 (20 -> 20 -> 20) + BN ----
  hipMemsetAsync(agg, 0, (size_t)N * 32 * sizeof(float), stream);
  scatter20_kernel<<<sc20_blocks, blk, 0, stream>>>(o1, ei, ew, agg, E);
  mlp_wmma_kernel<32, 1><<<mlp_blocks, blk, 0, stream>>>(agg, w1fb, bp + 64,
                                                         w2fb, bp + 96, o2, tiles);
  bn_stats_kernel<<<128, blk, 0, stream>>>(o2, stats + 64, stats + 96, N);
  bn_apply_kernel<<<elem_blocks, blk, 0, stream>>>(o2, stats + 64, stats + 96, g2, be2, N);

  // ---- layer 3 (20 -> 20 -> 20), no BN ----
  hipMemsetAsync(agg, 0, (size_t)N * 32 * sizeof(float), stream);
  scatter20_kernel<<<sc20_blocks, blk, 0, stream>>>(o2, ei, ew, agg, E);
  mlp_wmma_kernel<32, 1><<<mlp_blocks, blk, 0, stream>>>(agg, w1fc, bp + 128,
                                                         w2fc, bp + 160, o3, tiles);

  // ---- final linear ----
  final_linear_kernel<<<node_blocks, blk, 0, stream>>>(o1, o2, o3, wlin, blin,
                                                       (float*)d_out, N);
}
